// ValueNetworkGRU_22016002359301
// MI455X (gfx1250) — matrix operand order repaired and checked by reading, not compiled, not run on hardware
//
#include <hip/hip_runtime.h>

// ---------------------------------------------------------------------------
// Persistent batch-parallel GRU value network for MI455X (gfx1250, wave32).
// 16 workgroups (one 16-row batch tile each) x 512 threads (16 waves).
// GEMMs via v_wmma_f32_16x16x32_bf16; weights pre-converted to bf16 [N,K].
// x(t+1) tile prefetched with gfx1250 async global->LDS loads (ASYNCcnt).
// Weight pointers kept in address_space(1) and laundered per-step: loads are
// re-issued from L2 each iteration (no LICM hoist/spill) and stay GLOBAL
// encoding (FLAT would also consume DScnt and fight the LDS pipe).
// ---------------------------------------------------------------------------

typedef __attribute__((ext_vector_type(16))) __bf16    v16bf;
typedef __attribute__((ext_vector_type(8)))  float     v8f;
typedef __attribute__((ext_vector_type(4)))  int       v4i;
typedef __attribute__((ext_vector_type(4)))  unsigned  u32x4;

typedef __attribute__((address_space(1))) const unsigned short gus_t;
typedef __attribute__((address_space(1))) const u32x4          gu32x4;

union FragB16 { v16bf v; u32x4 q[2]; };

#if __has_builtin(__builtin_amdgcn_global_load_async_to_lds_b128)
#define HAS_ASYNC 1
#else
#define HAS_ASYNC 0
#endif

__device__ __forceinline__ void wait_async0() {
#if __has_builtin(__builtin_amdgcn_s_wait_asynccnt)
  __builtin_amdgcn_s_wait_asynccnt(0);
#else
  asm volatile("s_wait_asynccnt 0x0" ::: "memory");
#endif
}

#if HAS_ASYNC
typedef __attribute__((address_space(1))) v4i glb_v4i;
typedef __attribute__((address_space(3))) v4i lds_v4i;
__device__ __forceinline__ void async_ld_b128(const void* gsrc, void* ldst) {
  __builtin_amdgcn_global_load_async_to_lds_b128(
      (glb_v4i*)(gsrc), (lds_v4i*)(ldst), 0, 0);
}
#endif

__device__ __forceinline__ unsigned short f2bf(float f) {
  union { float f; unsigned u; } v; v.f = f;
  unsigned r = v.u + 0x7FFFu + ((v.u >> 16) & 1u);   // round-to-nearest-even
  return (unsigned short)(r >> 16);
}
__device__ __forceinline__ unsigned pack2bf(float a, float b) {
  return (unsigned)f2bf(a) | ((unsigned)f2bf(b) << 16);
}

__device__ __forceinline__ float fast_rcp(float x) {
#if __has_builtin(__builtin_amdgcn_rcpf)
  return __builtin_amdgcn_rcpf(x);
#else
  return 1.0f / x;
#endif
}
__device__ __forceinline__ float sigmoidf_(float x) { return fast_rcp(1.0f + __expf(-x)); }
__device__ __forceinline__ float tanhf_(float x)    { return 2.0f * sigmoidf_(2.0f * x) - 1.0f; }
__device__ __forceinline__ float lrelu_(float x)    { return x > 0.0f ? x : 0.01f * x; }

__device__ __forceinline__ v8f wmma_bf16(v16bf a, v16bf b, v8f c) {
  return __builtin_amdgcn_wmma_f32_16x16x32_bf16(false, a, false, b, (short)0, c,
                                                 false, false);
}

// A fragment (16x32 bf16, M x K): bf16 row-major [16][lda] in LDS.
// lanes 0-15 = M, hold K {kk..+7, kk+16..+23}; lanes 16-31 hold K {kk+8..+15, kk+24..+31}.
__device__ __forceinline__ v16bf load_a_lds(const unsigned short* A, int lda,
                                            int kk, int lane) {
  int m   = lane & 15;
  int off = (lane >> 4) << 3;
  FragB16 f;
  f.q[0] = *(const u32x4*)(A + m * lda + kk + off);
  f.q[1] = *(const u32x4*)(A + m * lda + kk + 16 + off);
  return f.v;
}

// B fragment (32x16 bf16, K x N) for W.T, W stored bf16 [N][ldb] row-major
// in global memory (address_space(1) => global_load_b128 encoding).
// lanes 0-15 = N hold K {kk..+7, kk+16..+23}; lanes 16-31 hold K {kk+8..+15, kk+24..+31}.
__device__ __forceinline__ v16bf load_b_gl(gus_t* W, int ldb, int nrow,
                                           int kk, int lane) {
  int n    = nrow + (lane & 15);
  int koff = kk + ((lane >> 4) << 3);
  gu32x4* p = (gu32x4*)(W + n * ldb + koff);
  FragB16 f;
  f.q[0] = p[0];
  f.q[1] = p[2];
  return f.v;
}

// ---------------------------------------------------------------------------
// Kernel 1: fp32 weights -> bf16 workspace (same [N,K] layouts).
//   [0,16384)        w_shared [256][64]
//   [16384,409600)   w_ih     [2][768][256]
//   [409600,802816)  w_hh     [2][768][256]
//   [802816,868352)  w_v1     [256][256]
// ---------------------------------------------------------------------------
__global__ void vng_convert_weights(const float* __restrict__ wsh,
                                    const float* __restrict__ wih,
                                    const float* __restrict__ whh,
                                    const float* __restrict__ wv1,
                                    unsigned short* __restrict__ ws) {
  int i = blockIdx.x * blockDim.x + threadIdx.x;
  if (i < 16384)            ws[i] = f2bf(wsh[i]);
  else if (i < 409600)      ws[i] = f2bf(wih[i - 16384]);
  else if (i < 802816)      ws[i] = f2bf(whh[i - 409600]);
  else if (i < 868352)      ws[i] = f2bf(wv1[i - 802816]);
}

// ---------------------------------------------------------------------------
// Kernel 2: persistent recurrent kernel.  Dynamic LDS carve (134144 B):
//   xb   : bf16 [16][64]     @ 0       x_t tile (bf16, GEMM A operand)
//   xin  : bf16 [16][256]    @ 2048    layer input / output
//   hbf  : bf16 [2][16][256] @ 10240   bf16 mirror of h
//   vred : f32  [256]        @ 26624   value reduction scratch
//   hf32 : f32  [2][16][256] @ 27648   fp32 master hidden state
//   G    : f32  [16][1024]   @ 60416   gate staging: r | z | i_n | h_n
//   xf32 : f32  [2][16][64]  @ 125952  async x double buffer
// ---------------------------------------------------------------------------
__global__ __launch_bounds__(512, 1) void vng_gru_persistent(
    const float* __restrict__ x,       // [512,256,64]
    const int*   __restrict__ dones,   // [512,256]
    const float* __restrict__ h0,      // [2,256,256]
    const float* __restrict__ b_shared,// [256]
    const float* __restrict__ b_ih,    // [2,768]
    const float* __restrict__ b_hh,    // [2,768]
    const float* __restrict__ b_v1,    // [256]
    const float* __restrict__ w_v2,    // [256]
    const float* __restrict__ b_v2,    // [1]
    const unsigned short* __restrict__ wsh0,  // bf16 [256][64]
    const unsigned short* __restrict__ wih0,  // bf16 [2][768][256]
    const unsigned short* __restrict__ whh0,  // bf16 [2][768][256]
    const unsigned short* __restrict__ wv10,  // bf16 [256][256]
    float* __restrict__ out)           // [512*256] v  then [2*256*256] h_final
{
  extern __shared__ char smem[];
  unsigned short* xb  = (unsigned short*)(smem);
  unsigned short* xin = (unsigned short*)(smem + 2048);
  unsigned short* hbf = (unsigned short*)(smem + 10240);
  float* vred = (float*)(smem + 26624);
  float* hf32 = (float*)(smem + 27648);
  float* G    = (float*)(smem + 60416);
  float* xf32 = (float*)(smem + 125952);

  const int tid  = threadIdx.x;
  const int lane = tid & 31;
  const int wave = tid >> 5;
  const int b0   = blockIdx.x * 16;   // batch tile base

  // ---- init hidden state from h0 ----
  for (int p = tid; p < 2 * 16 * 128; p += 512) {
    int idx = p << 1;
    int l = idx >> 12, m = (idx >> 8) & 15, j = idx & 255;
    float ha = h0[(l * 256 + (b0 + m)) * 256 + j];
    float hc = h0[(l * 256 + (b0 + m)) * 256 + j + 1];
    hf32[idx]     = ha;
    hf32[idx + 1] = hc;
    *(unsigned*)(hbf + idx) = pack2bf(ha, hc);
  }
#if HAS_ASYNC
  // prologue: async-load x(0) tile into xf32[0]
  if (tid < 256) {
    int m = tid >> 4, q = tid & 15;                 // 16B chunk per thread
    async_ld_b128(x + ((long)0 * 256 + b0 + m) * 64 + q * 4,
                  xf32 + m * 64 + q * 4);
  }
#endif
  __syncthreads();

  for (int t = 0; t < 512; ++t) {
    // Launder AS1 weight pointers (SGPR pair each): keeps loads inside the
    // time loop (re-read from L2) while preserving GLOBAL load encoding.
    gus_t* wsh = (gus_t*)wsh0;
    gus_t* wih = (gus_t*)wih0;
    gus_t* whh = (gus_t*)whh0;
    gus_t* wv1 = (gus_t*)wv10;
    asm("" : "+s"(wsh), "+s"(wih), "+s"(whh), "+s"(wv1));

    // ---- complete async x(t), then stage: convert x, mask h, kick x(t+1) ----
#if HAS_ASYNC
    wait_async0();
    __syncthreads();
    const float* xcur = xf32 + (t & 1) * 1024;
    for (int p = tid; p < 512; p += 512) {
      int idx = p << 1;
      *(unsigned*)(xb + idx) = pack2bf(xcur[idx], xcur[idx + 1]);
    }
    if ((t + 1 < 512) && tid < 256) {
      int m = tid >> 4, q = tid & 15;
      async_ld_b128(x + ((long)(t + 1) * 256 + b0 + m) * 64 + q * 4,
                    xf32 + ((t + 1) & 1) * 1024 + m * 64 + q * 4);
    }
#else
    for (int p = tid; p < 512; p += 512) {
      int idx = p << 1;
      int m = idx >> 6, k = idx & 63;
      const float* xr = x + ((long)t * 256 + b0 + m) * 64 + k;
      *(unsigned*)(xb + idx) = pack2bf(xr[0], xr[1]);
    }
    if (t + 1 < 512) {
      const char* nx = (const char*)(x + ((long)(t + 1) * 256 + b0 + (tid >> 5)) * 64);
      __builtin_prefetch(nx, 0, 1);
    }
#endif
    for (int p = tid; p < 2 * 16 * 128; p += 512) {
      int idx = p << 1;
      int m = (idx >> 8) & 15;
      float dm = 1.0f - (float)dones[t * 256 + b0 + m];
      float ha = hf32[idx] * dm;
      float hc = hf32[idx + 1] * dm;
      hf32[idx]     = ha;
      hf32[idx + 1] = hc;
      *(unsigned*)(hbf + idx) = pack2bf(ha, hc);
    }
    __syncthreads();

    // ---- feats = leaky_relu(x @ w_shared.T + b_shared) -> xin (bf16) ----
    {
      int nb   = wave * 16;
      int ncol = nb + (lane & 15);
      float bias = b_shared[ncol];
      v8f acc;
      #pragma unroll
      for (int r = 0; r < 8; ++r) acc[r] = bias;
      #pragma unroll
      for (int kk = 0; kk < 64; kk += 32) {
        v16bf a = load_a_lds(xb, 64, kk, lane);
        v16bf b = load_b_gl(wsh, 64, nb, kk, lane);
        acc = wmma_bf16(a, b, acc);
      }
      int mb = (lane >> 4) << 3;
      #pragma unroll
      for (int r = 0; r < 8; ++r)
        xin[(mb + r) * 256 + ncol] = f2bf(lrelu_(acc[r]));
    }
    __syncthreads();

    // ---- GRU layers ----
    for (int l = 0; l < 2; ++l) {
      gus_t* Wih = wih + l * 768 * 256;
      gus_t* Whh = whh + l * 768 * 256;
      const float* Bih = b_ih + l * 768;
      const float* Bhh = b_hh + l * 768;
      const unsigned short* hb = hbf + l * 4096;

      // 64 logical 16-wide tiles of G[16][1024]:
      //   tiles  0..31: r|z = x@Wih.T + h@Whh.T + bi + bh   (fused)
      //   tiles 32..47: i_n = x@Wih.T(rows 512..767) + bi
      //   tiles 48..63: h_n = h@Whh.T(rows 512..767) + bh
      for (int tt = 0; tt < 4; ++tt) {
        int tile = wave + tt * 16;
        int nb   = tile * 16;
        int lc   = lane & 15;
        int mb   = (lane >> 4) << 3;
        v8f acc;
        if (tile < 32) {
          float bias = Bih[nb + lc] + Bhh[nb + lc];
          #pragma unroll
          for (int r = 0; r < 8; ++r) acc[r] = bias;
          #pragma unroll
          for (int kk = 0; kk < 256; kk += 32) {
            v16bf ax = load_a_lds(xin, 256, kk, lane);
            v16bf bx = load_b_gl(Wih, 256, nb, kk, lane);
            acc = wmma_bf16(ax, bx, acc);
            v16bf ah   = load_a_lds(hb, 256, kk, lane);
            v16bf bh16 = load_b_gl(Whh, 256, nb, kk, lane);
            acc = wmma_bf16(ah, bh16, acc);
          }
        } else if (tile < 48) {
          int wn = nb;                       // 512..767
          float bias = Bih[wn + lc];
          #pragma unroll
          for (int r = 0; r < 8; ++r) acc[r] = bias;
          #pragma unroll
          for (int kk = 0; kk < 256; kk += 32) {
            v16bf a = load_a_lds(xin, 256, kk, lane);
            v16bf b = load_b_gl(Wih, 256, wn, kk, lane);
            acc = wmma_bf16(a, b, acc);
          }
        } else {
          int wn = nb - 256;                 // 512..767
          float bias = Bhh[wn + lc];
          #pragma unroll
          for (int r = 0; r < 8; ++r) acc[r] = bias;
          #pragma unroll
          for (int kk = 0; kk < 256; kk += 32) {
            v16bf a = load_a_lds(hb, 256, kk, lane);
            v16bf b = load_b_gl(Whh, 256, wn, kk, lane);
            acc = wmma_bf16(a, b, acc);
          }
        }
        #pragma unroll
        for (int r = 0; r < 8; ++r)
          G[(mb + r) * 1024 + nb + lc] = acc[r];
      }
      __syncthreads();

      // elementwise GRU update (fp32, TRANS-unit sigmoid/tanh), 2 j's/thread
      float* hf = hf32 + l * 4096;
      unsigned short* hbl = hbf + l * 4096;
      for (int p = tid; p < 16 * 128; p += 512) {
        int m = p >> 7, j = (p & 127) << 1;
        const float* gp = G + m * 1024;
        int base = m * 256 + j;
        float r0 = sigmoidf_(gp[j]);
        float r1 = sigmoidf_(gp[j + 1]);
        float z0 = sigmoidf_(gp[256 + j]);
        float z1 = sigmoidf_(gp[257 + j]);
        float n0 = tanhf_(gp[512 + j] + r0 * gp[768 + j]);
        float n1 = tanhf_(gp[513 + j] + r1 * gp[769 + j]);
        float ha = hf[base], hc = hf[base + 1];
        float h0n = (1.0f - z0) * n0 + z0 * ha;
        float h1n = (1.0f - z1) * n1 + z1 * hc;
        hf[base]     = h0n;
        hf[base + 1] = h1n;
        unsigned pk = pack2bf(h0n, h1n);
        *(unsigned*)(hbl + base) = pk;
        *(unsigned*)(xin + base) = pk;   // next layer's / value head's input
      }
      __syncthreads();
    }

    // ---- value head: v1 = leaky(gru_out @ w_v1.T + b_v1) -> G (fp32) ----
    {
      int nb   = wave * 16;
      int ncol = nb + (lane & 15);
      float bias = b_v1[ncol];
      v8f acc;
      #pragma unroll
      for (int r = 0; r < 8; ++r) acc[r] = bias;
      #pragma unroll
      for (int kk = 0; kk < 256; kk += 32) {
        v16bf a = load_a_lds(xin, 256, kk, lane);
        v16bf b = load_b_gl(wv1, 256, nb, kk, lane);
        acc = wmma_bf16(a, b, acc);
      }
      int mb = (lane >> 4) << 3;
      #pragma unroll
      for (int r = 0; r < 8; ++r)
        G[(mb + r) * 1024 + ncol] = lrelu_(acc[r]);
    }
    __syncthreads();

    // ---- v = v1 @ w_v2.T + b_v2 ----
    if (tid < 256) {
      int m = tid >> 4, gseg = tid & 15;
      float s = 0.0f;
      #pragma unroll
      for (int j = 0; j < 16; ++j)
        s += G[m * 1024 + gseg * 16 + j] * w_v2[gseg * 16 + j];
      vred[tid] = s;
    }
    __syncthreads();
    if (tid < 16) {
      float s = 0.0f;
      #pragma unroll
      for (int j = 0; j < 16; ++j) s += vred[tid * 16 + j];
      out[(long)t * 256 + b0 + tid] = s + b_v2[0];
    }
    __syncthreads();
  }

  // ---- final hidden state -> out[131072 + l*65536 + b*256 + j] ----
  for (int idx = tid; idx < 2 * 16 * 256; idx += 512) {
    int l = idx >> 12, m = (idx >> 8) & 15, j = idx & 255;
    out[131072 + (l * 256 + (b0 + m)) * 256 + j] = hf32[idx];
  }
}

// ---------------------------------------------------------------------------
extern "C" void kernel_launch(void* const* d_in, const int* in_sizes, int n_in,
                              void* d_out, int out_size, void* d_ws, size_t ws_size,
                              hipStream_t stream) {
  const float* x        = (const float*)d_in[0];
  const int*   dones    = (const int*)  d_in[1];
  const float* h0       = (const float*)d_in[2];
  const float* w_shared = (const float*)d_in[3];
  const float* b_shared = (const float*)d_in[4];
  const float* w_ih     = (const float*)d_in[5];
  const float* w_hh     = (const float*)d_in[6];
  const float* b_ih     = (const float*)d_in[7];
  const float* b_hh     = (const float*)d_in[8];
  const float* w_v1     = (const float*)d_in[9];
  const float* b_v1     = (const float*)d_in[10];
  const float* w_v2     = (const float*)d_in[11];
  const float* b_v2     = (const float*)d_in[12];

  unsigned short* ws = (unsigned short*)d_ws;   // 868352 bf16 elems = 1.7 MB

  vng_convert_weights<<<(868352 + 255) / 256, 256, 0, stream>>>(
      w_shared, w_ih, w_hh, w_v1, ws);

  vng_gru_persistent<<<16, 512, 134144, stream>>>(
      x, dones, h0, b_shared, b_ih, b_hh, b_v1, w_v2, b_v2,
      ws + 0,        // w_shared bf16
      ws + 16384,    // w_ih bf16
      ws + 409600,   // w_hh bf16
      ws + 802816,   // w_v1 bf16
      (float*)d_out);
}
